// TriModalCrossAttention_ver2_72919954752310
// MI455X (gfx1250) — matrix operand order repaired and checked by reading, not compiled, not run on hardware
//
#include <hip/hip_runtime.h>
#include <hip/hip_bf16.h>

#define BB 4
#define SS 2048
#define DD 1024
#define MM (BB * SS)          // 8192 rows per modality
#define MD ((long long)MM * DD)

typedef __attribute__((ext_vector_type(16))) __bf16        bf16x16;
typedef __attribute__((ext_vector_type(8)))  float         floatx8;
typedef __attribute__((ext_vector_type(4)))  float         floatx4;
typedef __attribute__((ext_vector_type(4)))  unsigned int  uintx4;
typedef __attribute__((ext_vector_type(2)))  unsigned int  uintx2;
typedef __attribute__((ext_vector_type(8)))  int           intx8;
typedef __attribute__((ext_vector_type(4)))  int           intx4;

__device__ __forceinline__ unsigned short f32_bf16(float f) {
  unsigned int u = __float_as_uint(f);
  u += 0x7FFFu + ((u >> 16) & 1u);   // round-to-nearest-even
  return (unsigned short)(u >> 16);
}

__device__ __forceinline__ floatx8 zero8() {
  floatx8 z;
  for (int i = 0; i < 8; ++i) z[i] = 0.0f;
  return z;
}

union FragCast { uintx4 u[2]; bf16x16 v; };

// A-matrix fragment 16x32 bf16 from row-major [row][k] LDS, stride in elems (mult of 8).
__device__ __forceinline__ bf16x16 load_frag_a(const unsigned short* p, int stride) {
  int lane = threadIdx.x & 31;
  int r  = lane & 15;
  int kb = (lane >> 4) << 3;
  FragCast f;
  f.u[0] = *(const uintx4*)(p + r * stride + kb);
  f.u[1] = *(const uintx4*)(p + r * stride + kb + 16);
  return f.v;
}

// B-matrix fragment 32x16 bf16 from row-major [k][n] LDS, stride in elems (mult of 8).
__device__ __forceinline__ bf16x16 load_frag_b(const unsigned short* p, int stride) {
  int lane = threadIdx.x & 31;
  int n  = lane & 15;
  int kb = (lane >> 4) << 3;
  FragCast f;
  unsigned short* e = (unsigned short*)&f;
  for (int s = 0; s < 8; ++s) {
    e[s]     = p[(kb + s) * stride + n];
    e[8 + s] = p[(kb + 16 + s) * stride + n];
  }
  return f.v;
}

// ---------------------------------------------------------------------------
// GEMM: out[z] = A[z] (M x K) @ (W + z*wstride) (K x N) [+ bias[z]]
// ---------------------------------------------------------------------------
#define ASTR 40
#define WSTR 136

template <typename OutT, bool A_BF16, bool ACCUM, bool BIAS>
__global__ __launch_bounds__(256) void gemm_kernel(
    const void* __restrict__ A0, const void* __restrict__ A1, const void* __restrict__ A2,
    const float* __restrict__ W, long long wstride,
    const float* __restrict__ bias,
    OutT* __restrict__ out) {
  __shared__ unsigned short As[128 * ASTR];
  __shared__ unsigned short Ws[32 * WSTR];

  const int z  = blockIdx.z;
  const int m0 = blockIdx.y * 128;
  const int n0 = blockIdx.x * 128;
  const void*  Av = (z == 0) ? A0 : (z == 1) ? A1 : A2;
  const float* Wm = W + (long long)z * wstride;
  const float* bm = bias ? (bias + (long long)z * DD) : nullptr;
  OutT*        om = out + (long long)z * MD;

  const int tid  = threadIdx.x;
  const int wid  = tid >> 5;
  const int lane = tid & 31;
  const int wm   = wid & 3;     // 4 M-groups * 32 rows
  const int wn   = wid >> 2;    // 2 N-groups * 64 cols
  const int hi   = lane >> 4;
  const int ln   = lane & 15;

  floatx8 acc[2][4];
  for (int i = 0; i < 2; ++i)
    for (int j = 0; j < 4; ++j) acc[i][j] = zero8();

  for (int kk = 0; kk < DD; kk += 32) {
    // Stage A tile 128x32 -> bf16 LDS
    {
      int row = tid >> 3;
      int kv  = (tid & 7) << 2;
      for (int p = 0; p < 4; ++p, row += 32) {
        uintx2 pk;
        if (A_BF16) {
          const unsigned short* a = (const unsigned short*)Av;
          pk = *(const uintx2*)(a + (long long)(m0 + row) * DD + kk + kv);
        } else {
          const float* a = (const float*)Av;
          floatx4 vv = *(const floatx4*)(a + (long long)(m0 + row) * DD + kk + kv);
          pk[0] = (unsigned)f32_bf16(vv[0]) | ((unsigned)f32_bf16(vv[1]) << 16);
          pk[1] = (unsigned)f32_bf16(vv[2]) | ((unsigned)f32_bf16(vv[3]) << 16);
        }
        *(uintx2*)(&As[row * ASTR + kv]) = pk;
      }
      if (kk + 32 < DD) {
        const char* pref = (const char*)Av +
            ((long long)(m0 + (tid >> 3)) * DD + kk + 32) * (A_BF16 ? 2 : 4);
        __builtin_prefetch(pref, 0, 0);
      }
    }
    // Stage W tile 32x128 -> bf16 LDS (natural [k][n] layout)
    {
      int idx = tid;
      for (int p = 0; p < 4; ++p, idx += 256) {
        int n4 = (idx & 31) << 2;
        int k  = idx >> 5;
        floatx4 vv = *(const floatx4*)(Wm + (long long)(kk + k) * DD + n0 + n4);
        uintx2 pk;
        pk[0] = (unsigned)f32_bf16(vv[0]) | ((unsigned)f32_bf16(vv[1]) << 16);
        pk[1] = (unsigned)f32_bf16(vv[2]) | ((unsigned)f32_bf16(vv[3]) << 16);
        *(uintx2*)(&Ws[k * WSTR + n4]) = pk;
      }
      if (kk + 32 < DD)
        __builtin_prefetch(Wm + (long long)(kk + 32 + (tid >> 5)) * DD + n0, 0, 0);
    }
    __syncthreads();

    bf16x16 af0 = load_frag_a(&As[(wm * 32 + 0)  * ASTR], ASTR);
    bf16x16 af1 = load_frag_a(&As[(wm * 32 + 16) * ASTR], ASTR);
    for (int nt = 0; nt < 4; ++nt) {
      bf16x16 bf = load_frag_b(&Ws[wn * 64 + nt * 16], WSTR);
      acc[0][nt] = __builtin_amdgcn_wmma_f32_16x16x32_bf16(
          false, af0, false, bf, (short)0, acc[0][nt], false, false);
      acc[1][nt] = __builtin_amdgcn_wmma_f32_16x16x32_bf16(
          false, af1, false, bf, (short)0, acc[1][nt], false, false);
    }
    __syncthreads();
  }

  for (int mt = 0; mt < 2; ++mt) {
    for (int nt = 0; nt < 4; ++nt) {
      int n = n0 + wn * 64 + nt * 16 + ln;
      float bv = BIAS ? bm[n] : 0.0f;
      for (int r = 0; r < 8; ++r) {
        int m = m0 + wm * 32 + mt * 16 + r + hi * 8;
        long long o = (long long)m * DD + n;
        float val = acc[mt][nt][r] + bv;
        if (sizeof(OutT) == 2) {
          om[o] = (OutT)f32_bf16(val);
        } else if (ACCUM) {
          om[o] = (OutT)((float)om[o] + val);
        } else {
          om[o] = (OutT)val;
        }
      }
    }
  }
}

// ---------------------------------------------------------------------------
// Flash attention with async-to-LDS (Q) and Tensor Data Mover (V tiles).
// ---------------------------------------------------------------------------
#define QSTR  1032   // Q   [32][1032]  bf16
#define KTSTR 72     // K^T [1024][72]  bf16 (d-major)
#define VSTR  1032   // V   [64][1032]  bf16 (key-major)
#define PSTR  72     // S/P [32][72]

__global__ __launch_bounds__(256) void attn_kernel(
    const unsigned short* __restrict__ Q,
    const unsigned short* __restrict__ K,
    const unsigned short* __restrict__ V,
    unsigned short* __restrict__ Ctx) {
  extern __shared__ char smem[];
  unsigned short* Qs  = (unsigned short*)smem;                               // 66048 B
  unsigned short* KVs = (unsigned short*)(smem + 66048);                     // 147456 B (K^T | V)
  float*          Ss  = (float*)(smem + 66048 + 147456);                     // 9216 B
  unsigned short* Ps  = (unsigned short*)(smem + 66048 + 147456 + 9216);     // 4608 B
  float*          alpha_s = (float*)(smem + 66048 + 147456 + 9216 + 4608);   // 32 f32
  float*          lsum_s  = alpha_s + 32;                                    // 32 f32

  const int mmod = blockIdx.z, b = blockIdx.y, rb = blockIdx.x;
  const int row0 = rb * 32;
  const long long qoff  = ((long long)(mmod * BB + b) * SS + row0) * DD;
  const long long kbase = (long long)(mmod * BB + b) * SS * DD;
  const long long vbase = (long long)(((mmod + 1) % 3) * BB + b) * SS * DD;
  const unsigned short* q = Q + qoff;
  const unsigned short* k = K + kbase;
  const unsigned short* v = V + vbase;

  const int tid = threadIdx.x, wid = tid >> 5, lane = tid & 31;
  const int wm = wid & 1, wn = wid >> 1;
  const int hi = lane >> 4, ln = lane & 15;
  const float scale = 0.03125f;  // 1/sqrt(1024)

  // Load Q block 32x1024 via async HBM->LDS (16 B per lane per issue, ASYNCcnt)
  {
    unsigned qs_base = (unsigned)(unsigned long long)(void*)Qs;
    for (int p = 0; p < 16; ++p) {
      int idx = p * 256 + tid;
      int row = idx >> 7;
      int d8  = (idx & 127) << 3;
      unsigned ldsa = qs_base + (unsigned)(row * QSTR + d8) * 2u;
      unsigned long long ga = (unsigned long long)(q + (long long)row * DD + d8);
      asm volatile("global_load_async_to_lds_b128 %0, %1, off"
                   :: "v"(ldsa), "v"(ga) : "memory");
    }
    asm volatile("s_wait_asynccnt 0x0" ::: "memory");
  }

  floatx8 oacc[16];
  for (int i = 0; i < 16; ++i) oacc[i] = zero8();
  float row_m = -3.0e38f, row_l = 0.0f;

  for (int kb = 0; kb < SS / 64; ++kb) {
    // Load K block transposed: KVs[d][key], d-major
    for (int p = 0; p < 32; ++p) {
      int idx = p * 256 + tid;
      int key = idx >> 7;
      int d8  = (idx & 127) << 3;
      uintx4 vv = *(const uintx4*)(k + (long long)(kb * 64 + key) * DD + d8);
      const unsigned short* e = (const unsigned short*)&vv;
      for (int j = 0; j < 8; ++j) KVs[(d8 + j) * KTSTR + key] = e[j];
    }
    __syncthreads();

    // Scores: each wave one 16x16 tile of the 32x64 block
    {
      floatx8 sacc = zero8();
      for (int ks = 0; ks < DD / 32; ++ks) {
        bf16x16 af = load_frag_a(&Qs[wm * 16 * QSTR + ks * 32], QSTR);
        bf16x16 bf = load_frag_b(&KVs[(ks * 32) * KTSTR + wn * 16], KTSTR);
        sacc = __builtin_amdgcn_wmma_f32_16x16x32_bf16(
            false, af, false, bf, (short)0, sacc, false, false);
      }
      for (int r = 0; r < 8; ++r) {
        int row = wm * 16 + r + hi * 8;
        int col = wn * 16 + ln;
        Ss[row * PSTR + col] = sacc[r] * scale;
      }
    }
    __syncthreads();

    // Online softmax on wave 0 (one row per lane)
    if (tid < 32) {
      int row = tid;
      float mx = row_m;
      for (int j = 0; j < 64; ++j) mx = fmaxf(mx, Ss[row * PSTR + j]);
      float al = __expf(row_m - mx);
      float sum = 0.0f;
      for (int j = 0; j < 64; ++j) {
        float pv = __expf(Ss[row * PSTR + j] - mx);
        sum += pv;
        Ps[row * PSTR + j] = f32_bf16(pv);
      }
      row_m = mx;
      row_l = row_l * al + sum;
      alpha_s[row] = al;
    }
    __syncthreads();

    // Rescale O accumulators
    for (int r = 0; r < 8; ++r) {
      float al = alpha_s[wm * 16 + r + hi * 8];
      for (int dt = 0; dt < 16; ++dt) oacc[dt][r] *= al;
    }

    // V tile (64 x 1024 bf16) via Tensor Data Mover, iterate mode:
    // 64 iterations of a 1x1024 row; LDS stride padded to 1032 elems by the DMA.
    if (wid == 0) {
      unsigned long long ga = (unsigned long long)(v + (long long)kb * 64 * DD);
      unsigned ldsa = (unsigned)(unsigned long long)(void*)KVs;
      uintx4 g0;
      g0[0] = 1u;                                               // count=1, user mode
      g0[1] = ldsa;                                             // lds_addr (bytes)
      g0[2] = (unsigned)(ga & 0xFFFFFFFFull);                   // global_addr[31:0]
      g0[3] = (unsigned)((ga >> 32) & 0x01FFFFFFull) | (2u << 30);  // addr[56:32] | type=2
      intx8 g1;
      g1[0] = (int)((1u << 16) | (1u << 19));  // data_size=2B, iterate_enable=1
      g1[1] = (int)(1024u << 16);              // tensor_dim0[15:0] = 1024
      g1[2] = (int)(64u << 16);                // tensor_dim0 hi=0, tensor_dim1 lo = 64
      g1[3] = (int)(1024u << 16);              // tensor_dim1 hi=0, tile_dim0 = 1024
      g1[4] = 1;                               // tile_dim1 = 1, tile_dim2 = 0
      g1[5] = 1024;                            // tensor_dim0_stride lo
      g1[6] = (int)(1024u << 16);              // stride hi=0, tensor_dim1_stride lo
      g1[7] = 0;
      intx4 g2;
      g2[0] = 1;                               // tensor_dim2 (unused in iterate mode)
      g2[1] = 1032;                            // lds_addr_increment (data_size units)
      g2[2] = 1024;                            // global_addr_increment lo
      g2[3] = (int)(63u << 16);                // inc hi=0, iterate_count = 63 (64x)
      intx4 g3;
      g3[0] = 0; g3[1] = 0; g3[2] = 0; g3[3] = 0;
      intx8 g4;                                // extended group (6-arg builtin), unused
      for (int i = 0; i < 8; ++i) g4[i] = 0;
      __builtin_amdgcn_tensor_load_to_lds(g0, g1, g2, g3, g4, 0);
      __builtin_amdgcn_s_wait_tensorcnt(0);
    }
    __syncthreads();

    // O += P @ V  (wave: rows wm*16, d-cols wn*256 .. +255)
    for (int ks = 0; ks < 2; ++ks) {
      bf16x16 pf = load_frag_a(&Ps[wm * 16 * PSTR + ks * 32], PSTR);
      for (int dt = 0; dt < 16; ++dt) {
        bf16x16 vf = load_frag_b(&KVs[(ks * 32) * VSTR + wn * 256 + dt * 16], VSTR);
        oacc[dt] = __builtin_amdgcn_wmma_f32_16x16x32_bf16(
            false, pf, false, vf, (short)0, oacc[dt], false, false);
      }
    }
    __syncthreads();
  }

  if (tid < 32) lsum_s[tid] = row_l;
  __syncthreads();

  unsigned short* co = Ctx + qoff;
  for (int r = 0; r < 8; ++r) {
    int row = wm * 16 + r + hi * 8;
    float inv = 1.0f / lsum_s[row];
    for (int dt = 0; dt < 16; ++dt) {
      int col = wn * 256 + dt * 16 + ln;
      co[(long long)row * DD + col] = f32_bf16(oacc[dt][r] * inv);
    }
  }
}

// global_feature[b, m*S+s, d] = out_m[b, s, d]
__global__ __launch_bounds__(256) void concat_kernel(const float* __restrict__ src,
                                                     float* __restrict__ dst) {
  long long i4   = (long long)blockIdx.x * 256 + threadIdx.x;
  long long flat = i4 * 4;
  int d  = (int)(flat % DD);
  long long t  = flat / DD;
  int s  = (int)(t % SS);
  long long mb = t / SS;
  int mm = (int)(mb / BB);
  int bb = (int)(mb % BB);
  long long di = ((long long)(bb * 3 + mm) * SS + s) * DD + d;
  *(floatx4*)(dst + di) = *(const floatx4*)(src + flat);
}

extern "C" void kernel_launch(void* const* d_in, const int* in_sizes, int n_in,
                              void* d_out, int out_size, void* d_ws, size_t ws_size,
                              hipStream_t stream) {
  const float* x1 = (const float*)d_in[0];
  const float* x2 = (const float*)d_in[1];
  const float* x3 = (const float*)d_in[2];
  const float* Wq = (const float*)d_in[3];
  const float* bq = (const float*)d_in[4];
  const float* Wk = (const float*)d_in[5];
  const float* bk = (const float*)d_in[6];
  const float* Wv = (const float*)d_in[7];
  const float* bv = (const float*)d_in[8];
  const float* Wo = (const float*)d_in[9];
  const float* bo = (const float*)d_in[10];
  float* out = (float*)d_out;

  // Workspace (bf16): q | k | v | ctx, each 3*M*D elems = 50.3 MB (201 MB total)
  unsigned short* ws = (unsigned short*)d_ws;
  const long long QB = 3LL * MD;
  unsigned short* qw = ws;
  unsigned short* kw = ws + QB;
  unsigned short* vw = ws + 2 * QB;
  unsigned short* cw = ws + 3 * QB;

  dim3 blk(256);
  dim3 gg(DD / 128, MM / 128, 3);  // (8, 64, 3)

  // QKV projections -> bf16
  gemm_kernel<unsigned short, false, false, true><<<gg, blk, 0, stream>>>(
      x1, x2, x3, Wq, (long long)DD * DD, bq, qw);
  gemm_kernel<unsigned short, false, false, true><<<gg, blk, 0, stream>>>(
      x1, x2, x3, Wk, (long long)DD * DD, bk, kw);
  gemm_kernel<unsigned short, false, false, true><<<gg, blk, 0, stream>>>(
      x1, x2, x3, Wv, (long long)DD * DD, bv, vw);

  // Flash attention (v rolled by one modality inside the kernel)
  dim3 ga(SS / 32, BB, 3);
  size_t smem = 66048 + 147456 + 9216 + 4608 + 256;  // ~222 KB of the 320 KB WGP LDS
  attn_kernel<<<ga, blk, smem, stream>>>(qw, kw, vw, cw);

  // out_m = x_m @ Wo_m[:D] + bo_m
  gemm_kernel<float, false, false, true><<<gg, blk, 0, stream>>>(
      x1, x2, x3, Wo, 2LL * DD * DD, bo, out);
  // out_m += ctx_m @ Wo_m[D:]
  gemm_kernel<float, true, true, false><<<gg, blk, 0, stream>>>(
      cw, cw + MD, cw + 2 * MD, Wo + (long long)DD * DD, 2LL * DD * DD, nullptr, out);

  // global_feature
  concat_kernel<<<dim3((unsigned)(3 * MD / 1024)), blk, 0, stream>>>(out, out + 3 * MD);
}
